// SpatialAttention_57251914056356
// MI455X (gfx1250) — matrix operand order repaired and checked by reading, not compiled, not run on hardware
//
#include <hip/hip_runtime.h>

typedef unsigned short ushort_t;
typedef __bf16 bf16_t;
typedef bf16_t v16bf __attribute__((ext_vector_type(16)));
typedef float v8f __attribute__((ext_vector_type(8)));
typedef int int4v __attribute__((ext_vector_type(4)));

#define GK 768     // K for all dense GEMMs
#define GN 768
#define NTOK 196
#define KPAD 224   // keys padded to multiple of 32

#if defined(__gfx1250__) && __has_builtin(__builtin_amdgcn_global_load_async_to_lds_b128)
#define HAVE_ASYNC 1
#else
#define HAVE_ASYNC 0
#endif

static __device__ __forceinline__ ushort_t f2bf(float f) {
  unsigned u = __float_as_uint(f);
  u += 0x7FFFu + ((u >> 16) & 1u);   // round to nearest even
  return (ushort_t)(u >> 16);
}

static __device__ __forceinline__ v16bf frag_from(uint4 lo, uint4 hi) {
  union { uint4 u[2]; v16bf v; } c;
  c.u[0] = lo; c.u[1] = hi;
  return c.v;
}

static __device__ __forceinline__ v8f vzero8() {
  v8f z;
#pragma unroll
  for (int i = 0; i < 8; ++i) z[i] = 0.0f;
  return z;
}

#if HAVE_ASYNC
static __device__ __forceinline__ void async_ld_b128(const void* g, void* l) {
  __builtin_amdgcn_global_load_async_to_lds_b128(
      (__attribute__((address_space(1))) int4v*)(g),
      (__attribute__((address_space(3))) int4v*)(l), 0, 0);
}
static __device__ __forceinline__ void wait_async0() {
#if __has_builtin(__builtin_amdgcn_s_wait_asynccnt)
  __builtin_amdgcn_s_wait_asynccnt(0);
#else
  asm volatile("s_wait_asynccnt 0x0" ::: "memory");
#endif
}
#endif

// ---------------------------------------------------------------- f32 -> bf16
// vectorized: float4 -> 4 packed bf16 (one b64 store); n must be % 4 == 0
__global__ void cvt_bf16_kernel(const float* __restrict__ src,
                                ushort_t* __restrict__ dst, int n4) {
  for (int i = blockIdx.x * blockDim.x + threadIdx.x; i < n4;
       i += gridDim.x * blockDim.x) {
    float4 f = ((const float4*)src)[i];
    uint2 p;
    p.x = (unsigned)f2bf(f.x) | ((unsigned)f2bf(f.y) << 16);
    p.y = (unsigned)f2bf(f.z) | ((unsigned)f2bf(f.w) << 16);
    ((uint2*)dst)[i] = p;
  }
}

// ---------------------------------------------------------------- GEMM
// C[m,n] = sum_k A[m,k] * W[n,k] + bias[n]
// MODE 0: scatter bf16 to [B,H,N,D]  (q/k/v)
// MODE 1: f32 row-major [M,768]      (final output)
template <int MODE>
__global__ __launch_bounds__(256) void gemm_bias_kernel(
    const ushort_t* __restrict__ A, const ushort_t* __restrict__ W,
    const float* __restrict__ bias, ushort_t* __restrict__ outb,
    float* __restrict__ outf) {
  __shared__ uint4 As4[2][512];  // 128 rows x 32 bf16, XOR-swizzled 16B blocks
  __shared__ uint4 Bs4[2][512];  // 128 rows (n) x 32 bf16 (k)

  const int tid  = threadIdx.x;
  const int lane = tid & 31;
  const int wid  = tid >> 5;
  const int wm   = wid & 3;   // 4 wave-rows  -> 32 rows each
  const int wn   = wid >> 2;  // 2 wave-cols  -> 64 cols each
  const int m0   = blockIdx.y * 128;
  const int n0   = blockIdx.x * 128;
  const int l15  = lane & 15;
  const int half = lane >> 4;

  v8f acc[2][4];
#pragma unroll
  for (int i = 0; i < 2; ++i)
#pragma unroll
    for (int j = 0; j < 4; ++j) acc[i][j] = vzero8();

#if HAVE_ASYNC
  // issue async copy of one 128x32 A tile + 128x32 B tile into LDS buffer `buf`
  auto issue_tile = [&](int k0, int buf) {
#pragma unroll
    for (int r = 0; r < 2; ++r) {
      int idx = tid + r * 256;             // 0..511
      int row = idx >> 2;
      int cb  = idx & 3;
      const ushort_t* ga = A + (size_t)(m0 + row) * GK + k0 + cb * 8;
      const ushort_t* gb = W + (size_t)(n0 + row) * GK + k0 + cb * 8;
      async_ld_b128(ga, &As4[buf][row * 4 + (cb ^ (row & 3))]);
      async_ld_b128(gb, &Bs4[buf][row * 4 + (cb ^ (row & 3))]);
    }
  };
  issue_tile(0, 0);
  wait_async0();
  __syncthreads();
#else
  uint4 pa[2], pb[2];
  auto fetch_tile = [&](int k0) {
#pragma unroll
    for (int r = 0; r < 2; ++r) {
      int idx = tid + r * 256;
      int row = idx >> 2;
      int cb  = idx & 3;
      pa[r] = *(const uint4*)(A + (size_t)(m0 + row) * GK + k0 + cb * 8);
      pb[r] = *(const uint4*)(W + (size_t)(n0 + row) * GK + k0 + cb * 8);
    }
  };
  fetch_tile(0);
#endif

  for (int kt = 0; kt < GK / 32; ++kt) {
#if HAVE_ASYNC
    const int cur = kt & 1;
    if (kt + 1 < GK / 32) issue_tile((kt + 1) * 32, cur ^ 1);
#else
    const int cur = 0;
    __syncthreads();
#pragma unroll
    for (int r = 0; r < 2; ++r) {
      int idx = tid + r * 256;
      int row = idx >> 2;
      int cb  = idx & 3;
      As4[0][row * 4 + (cb ^ (row & 3))] = pa[r];
      Bs4[0][row * 4 + (cb ^ (row & 3))] = pb[r];
    }
    __syncthreads();
    if (kt + 1 < GK / 32) fetch_tile((kt + 1) * 32);
#endif

    v16bf af[2], bfr[4];
#pragma unroll
    for (int ti = 0; ti < 2; ++ti) {
      int r = wm * 32 + ti * 16 + l15;
      // A 16-bit layout: lane-half 0 -> K 0-7 & 16-23; half 1 -> K 8-15 & 24-31
      uint4 lo = As4[cur][r * 4 + ((half)     ^ (r & 3))];
      uint4 hi = As4[cur][r * 4 + ((2 + half) ^ (r & 3))];
      af[ti] = frag_from(lo, hi);
    }
#pragma unroll
    for (int tj = 0; tj < 4; ++tj) {
      int r = wn * 64 + tj * 16 + l15;     // r = output column (n)
      // B 16-bit layout: lane = column, lane-half 0 -> K 0-7 & 16-23; half 1 -> K 8-15 & 24-31
      uint4 lo = Bs4[cur][r * 4 + ((half)     ^ (r & 3))];
      uint4 hi = Bs4[cur][r * 4 + ((2 + half) ^ (r & 3))];
      bfr[tj] = frag_from(lo, hi);
    }
#pragma unroll
    for (int ti = 0; ti < 2; ++ti)
#pragma unroll
      for (int tj = 0; tj < 4; ++tj)
        acc[ti][tj] = __builtin_amdgcn_wmma_f32_16x16x32_bf16(
            false, af[ti], false, bfr[tj], (short)0, acc[ti][tj], false, false);

#if HAVE_ASYNC
    wait_async0();
    __syncthreads();
#endif
  }

#pragma unroll
  for (int ti = 0; ti < 2; ++ti) {
#pragma unroll
    for (int tj = 0; tj < 4; ++tj) {
      int n = n0 + wn * 64 + tj * 16 + l15;
      float bv = bias[n];
#pragma unroll
      for (int e = 0; e < 8; ++e) {
        int m = m0 + wm * 32 + ti * 16 + e + 8 * half;
        float val = acc[ti][tj][e] + bv;
        if (MODE == 0) {
          int btok = m / NTOK, ntok = m - btok * NTOK;
          int hh = n >> 6, d = n & 63;
          outb[(((size_t)btok * 12 + hh) * NTOK + ntok) * 64 + d] = f2bf(val);
        } else {
          outf[(size_t)m * GN + n] = val;
        }
      }
    }
  }
}

// ---------------------------------------------------------------- attention
// per (b,h): S = qk^T/8 + bias, masked softmax, O = P v ; O -> [B,N,768] bf16
__global__ __launch_bounds__(128) void attn_kernel(
    const ushort_t* __restrict__ Q, const ushort_t* __restrict__ Kb,
    const ushort_t* __restrict__ V, const float* __restrict__ pos_bias,
    const int* __restrict__ mask, ushort_t* __restrict__ AO) {
  __shared__ ushort_t vT[64 * KPAD];       // V transposed: [d][key]
  __shared__ ushort_t Pb[4 * 16 * KPAD];   // per-wave probability rows

  const int bh   = blockIdx.x;
  const int b    = bh / 12;
  const int h    = bh - b * 12;
  const int tid  = threadIdx.x;
  const int lane = tid & 31;
  const int wid  = tid >> 5;
  const int l15  = lane & 15;
  const int half = lane >> 4;
  const size_t base = (size_t)bh * NTOK * 64;

  for (int e = tid; e < NTOK * 64; e += 128) {
    int key = e >> 6, d = e & 63;
    vT[d * KPAD + key] = V[base + e];
  }
  for (int e = tid; e < (KPAD - NTOK) * 64; e += 128) {
    int d = e / (KPAD - NTOK);
    int key = NTOK + (e - d * (KPAD - NTOK));
    vT[d * KPAD + key] = 0;
  }
  __syncthreads();

  // per-column additive penalty (0 if valid, -1e30 if masked / padded col),
  // hoisted out of the query-tile loop; keeps softmax math branchless
  float colpen[13];
#pragma unroll
  for (int kt = 0; kt < 13; ++kt) {
    int col = kt * 16 + l15;
    int cc  = col < NTOK ? col : NTOK - 1;     // clamped, always in-bounds
    bool ok = (col < NTOK) & (mask[b * NTOK + cc] != 0);
    colpen[kt] = ok ? 0.0f : -1.0e30f;
  }

  ushort_t* Pw = Pb + wid * 16 * KPAD;

  for (int qt = wid; qt < 13; qt += 4) {
    int qrow = qt * 16 + l15;
    int qr = qrow < NTOK ? qrow : NTOK - 1;   // clamp; padded rows discarded

    v16bf qf[2];
#pragma unroll
    for (int s = 0; s < 2; ++s) {
      const ushort_t* p = Q + base + (size_t)qr * 64 + s * 32 + half * 8;
      qf[s] = frag_from(*(const uint4*)p, *(const uint4*)(p + 16));
    }

    v8f sacc[13];
#pragma unroll
    for (int kt = 0; kt < 13; ++kt) {
      v8f z = vzero8();
      int keyr = kt * 16 + l15;
      int kr = keyr < NTOK ? keyr : NTOK - 1; // clamp; padded cols masked later
#pragma unroll
      for (int s = 0; s < 2; ++s) {
        const ushort_t* p = Kb + base + (size_t)kr * 64 + s * 32 + half * 8;
        v16bf kf = frag_from(*(const uint4*)p, *(const uint4*)(p + 16));
        z = __builtin_amdgcn_wmma_f32_16x16x32_bf16(
            false, qf[s], false, kf, (short)0, z, false, false);
      }
      sacc[kt] = z;
    }

    // softmax: element (row = qt*16 + e + 8*half, col = kt*16 + l15)
#pragma unroll
    for (int e = 0; e < 8; ++e) {
      int row = qt * 16 + e + 8 * half;
      int rc  = row < NTOK ? row : NTOK - 1;  // clamped row for bias loads
      float rowpen = (row < NTOK) ? 0.0f : -1.0e30f;
      float mx = -3.0e38f;
#pragma unroll
      for (int kt = 0; kt < 13; ++kt) {
        int col = kt * 16 + l15;
        int cc  = col < NTOK ? col : NTOK - 1;
        // unconditional clamped load; additive penalty keeps the load and the
        // fma live on every path (nothing for the compiler to sink/branch)
        float pb = pos_bias[rc * NTOK + cc];
        float v = sacc[kt][e] * 0.125f + pb + (rowpen + colpen[kt]);
        sacc[kt][e] = v;
        mx = fmaxf(mx, v);
      }
#pragma unroll
      for (int off = 1; off < 16; off <<= 1)
        mx = fmaxf(mx, __shfl_xor(mx, off, 32));
      float sum = 0.0f;
#pragma unroll
      for (int kt = 0; kt < 13; ++kt) {
        float p = __expf(sacc[kt][e] - mx);
        sacc[kt][e] = p;
        sum += p;
      }
#pragma unroll
      for (int off = 1; off < 16; off <<= 1) sum += __shfl_xor(sum, off, 32);
      float inv = 1.0f / sum;
#pragma unroll
      for (int kt = 0; kt < 13; ++kt)
        Pw[(e + 8 * half) * KPAD + kt * 16 + l15] = f2bf(sacc[kt][e] * inv);
      Pw[(e + 8 * half) * KPAD + 208 + l15] = 0;   // zero tail key tile
    }
    // same-wave LDS ops are in-order: Pw write->read hazard is safe

    v8f oacc[4];
#pragma unroll
    for (int nt = 0; nt < 4; ++nt) oacc[nt] = vzero8();
#pragma unroll
    for (int ks = 0; ks < 7; ++ks) {
      int k0 = ks * 32;
      const ushort_t* pa = Pw + l15 * KPAD + k0 + half * 8;
      v16bf af = frag_from(*(const uint4*)pa, *(const uint4*)(pa + 16));
#pragma unroll
      for (int nt = 0; nt < 4; ++nt) {
        const ushort_t* pb = vT + (nt * 16 + l15) * KPAD + k0 + half * 8;
        v16bf bfv = frag_from(*(const uint4*)pb, *(const uint4*)(pb + 16));
        oacc[nt] = __builtin_amdgcn_wmma_f32_16x16x32_bf16(
            false, af, false, bfv, (short)0, oacc[nt], false, false);
      }
    }

#pragma unroll
    for (int nt = 0; nt < 4; ++nt) {
#pragma unroll
      for (int e = 0; e < 8; ++e) {
        int row = qt * 16 + e + 8 * half;
        if (row < NTOK) {
          int col = h * 64 + nt * 16 + l15;
          AO[((size_t)b * NTOK + row) * GN + col] = f2bf(oacc[nt][e]);
        }
      }
    }
  }
}

// ---------------------------------------------------------------- launcher
extern "C" void kernel_launch(void* const* d_in, const int* in_sizes, int n_in,
                              void* d_out, int out_size, void* d_ws,
                              size_t ws_size, hipStream_t stream) {
  const float* x   = (const float*)d_in[0];
  const float* Wq  = (const float*)d_in[1];
  const float* bq  = (const float*)d_in[2];
  const float* Wk  = (const float*)d_in[3];
  const float* bk  = (const float*)d_in[4];
  const float* Wv  = (const float*)d_in[5];
  const float* bv  = (const float*)d_in[6];
  const float* Wp  = (const float*)d_in[7];
  const float* bp  = (const float*)d_in[8];
  const float* pos = (const float*)d_in[9];
  const int*   msk = (const int*)d_in[10];

  const size_t M = 128 * 196;           // 25088
  const size_t SZ_X = M * GK * 2;       // bf16 activations
  const size_t SZ_W = (size_t)GN * GK * 2;

  char* ws = (char*)d_ws;
  ushort_t* XB  = (ushort_t*)(ws);
  ushort_t* WQB = (ushort_t*)(ws + SZ_X);
  ushort_t* WKB = (ushort_t*)(ws + SZ_X + SZ_W);
  ushort_t* WVB = (ushort_t*)(ws + SZ_X + 2 * SZ_W);
  ushort_t* WPB = (ushort_t*)(ws + SZ_X + 3 * SZ_W);
  ushort_t* QB  = (ushort_t*)(ws + SZ_X + 4 * SZ_W);
  ushort_t* KB  = (ushort_t*)(ws + 2 * SZ_X + 4 * SZ_W);
  ushort_t* VB  = (ushort_t*)(ws + 3 * SZ_X + 4 * SZ_W);
  ushort_t* AOB = (ushort_t*)(ws + 4 * SZ_X + 4 * SZ_W);

  cvt_bf16_kernel<<<1024, 256, 0, stream>>>(x, XB, (int)(M * GK / 4));
  cvt_bf16_kernel<<<256, 256, 0, stream>>>(Wq, WQB, GN * GK / 4);
  cvt_bf16_kernel<<<256, 256, 0, stream>>>(Wk, WKB, GN * GK / 4);
  cvt_bf16_kernel<<<256, 256, 0, stream>>>(Wv, WVB, GN * GK / 4);
  cvt_bf16_kernel<<<256, 256, 0, stream>>>(Wp, WPB, GN * GK / 4);

  dim3 ggrid(GN / 128, (unsigned)(M / 128));
  gemm_bias_kernel<0><<<ggrid, 256, 0, stream>>>(XB, WQB, bq, QB, nullptr);
  gemm_bias_kernel<0><<<ggrid, 256, 0, stream>>>(XB, WKB, bk, KB, nullptr);
  gemm_bias_kernel<0><<<ggrid, 256, 0, stream>>>(XB, WVB, bv, VB, nullptr);

  attn_kernel<<<128 * 12, 128, 0, stream>>>(QB, KB, VB, pos, msk, AOB);

  gemm_bias_kernel<1><<<ggrid, 256, 0, stream>>>(AOB, WPB, bp, nullptr,
                                                 (float*)d_out);
}
